// AttentionBlock_11467562680975
// MI455X (gfx1250) — matrix operand order repaired and checked by reading, not compiled, not run on hardware
//
#include <hip/hip_runtime.h>
#include <hip/hip_bf16.h>
#include <math.h>

// Problem constants (from reference)
#define B_  4
#define S_  2048
#define D_  1024
#define H_  16
#define DK_ 64

typedef __bf16 bf16;
typedef __attribute__((ext_vector_type(16))) __bf16 v16bf;
typedef __attribute__((ext_vector_type(8)))  __bf16 v8bf;
typedef __attribute__((ext_vector_type(8)))  float  v8f;

// ---------------------------------------------------------------------------
// WMMA fragment loaders (CDNA5 16-bit layouts, wave32)
//
// A-matrix 16x32 (MxK): lane L (0..15) = row M=L, lanes 16..31 = row M=L-16.
//   halfword e in [0,8):  K = k0 + 8*hi + e          (contiguous 8 elements)
//   halfword e in [8,16): K = k0 + 16 + 8*hi + (e-8) (contiguous 8 elements)
// B-matrix 32x16 (KxN): lane L (0..15) = col N=L; halfword e: same K mapping as A
//   -> two contiguous runs of 8 bf16 (16 bytes) per lane.
// ---------------------------------------------------------------------------
static __device__ __forceinline__ v16bf load_a_frag(const bf16* rowbase, int k0, int hi) {
    const v8bf lo = *(const v8bf*)(rowbase + k0 + 8 * hi);
    const v8bf hc = *(const v8bf*)(rowbase + k0 + 16 + 8 * hi);
    v16bf a;
#pragma unroll
    for (int i = 0; i < 8; ++i) { a[i] = lo[i]; a[i + 8] = hc[i]; }
    return a;
}

static __device__ __forceinline__ v16bf load_b_frag(const bf16* rowbase, int k0, int hi) {
    const v8bf lo = *(const v8bf*)(rowbase + k0 + 8 * hi);
    const v8bf hc = *(const v8bf*)(rowbase + k0 + 16 + 8 * hi);
    v16bf b;
#pragma unroll
    for (int i = 0; i < 8; ++i) { b[i] = lo[i]; b[i + 8] = hc[i]; }
    return b;
}

static __device__ __forceinline__ v8f wmma_bf16(v16bf a, v16bf b, v8f c) {
    return __builtin_amdgcn_wmma_f32_16x16x32_bf16(false, a, false, b, (short)0, c, false, false);
}

// ---------------------------------------------------------------------------
// Elementwise conversion fp32 -> bf16
// ---------------------------------------------------------------------------
__global__ void cvt_f32_bf16(const float* __restrict__ x, bf16* __restrict__ y, size_t n) {
    size_t i = (size_t)blockIdx.x * blockDim.x + threadIdx.x;
    const size_t stride = (size_t)gridDim.x * blockDim.x;
    for (; i < n; i += stride) y[i] = (bf16)x[i];
}

// Wt[n*Dim + k] = bf16(W[k*Dim + n])  (transpose + convert; coalesced writes)
__global__ void transpose_w_bf16(const float* __restrict__ W, bf16* __restrict__ Wt, int Dim) {
    size_t i = (size_t)blockIdx.x * blockDim.x + threadIdx.x;
    const size_t stride = (size_t)gridDim.x * blockDim.x;
    const size_t total = (size_t)Dim * Dim;
    for (; i < total; i += stride) {
        const int n = (int)(i / Dim);
        const int k = (int)(i % Dim);
        Wt[i] = (bf16)W[(size_t)k * Dim + n];
    }
}

// Vt[((b*H+h)*DK + dk)*S + s] = Vb[(b*S+s)*D + h*DK + dk]
__global__ void v_transpose(const bf16* __restrict__ Vb, bf16* __restrict__ Vt) {
    size_t i = (size_t)blockIdx.x * blockDim.x + threadIdx.x;
    const size_t stride = (size_t)gridDim.x * blockDim.x;
    const size_t total = (size_t)B_ * S_ * D_;
    for (; i < total; i += stride) {
        const int b  = (int)(i / ((size_t)S_ * D_));
        const int sr = (int)(i % ((size_t)S_ * D_));
        const int s  = sr / D_;
        const int c  = sr % D_;
        const int h  = c / DK_;
        const int dk = c % DK_;
        Vt[(((size_t)b * H_ + h) * DK_ + dk) * S_ + s] = Vb[i];
    }
}

// ---------------------------------------------------------------------------
// Generic bf16 GEMM:  out = (A[M,K] @ Wt[N,K]^T + bias[N]) * alpha
// One wave computes a 32x32 output tile (2 A-frags x 2 B-frags -> 4 WMMA per
// k-step, 4 f32 accumulators). Register-level reuse: each fragment load feeds
// two WMMAs, halving bytes/FLOP vs a 16x16 wave tile. Block = 32x4 (4 waves).
// ---------------------------------------------------------------------------
__global__ void gemm_bf16(const bf16* __restrict__ A, const bf16* __restrict__ Wt,
                          const float* __restrict__ bias, float alpha,
                          bf16* __restrict__ outb, float* __restrict__ outf,
                          int M, int N, int K) {
    const int lane = threadIdx.x;
    const int ln   = lane & 15;
    const int hi   = lane >> 4;
    const int n0   = blockIdx.x * 32;
    const int m0   = (blockIdx.y * blockDim.y + threadIdx.y) * 32;
    if (m0 >= M) return;

    const bf16* arow0 = A  + (size_t)(m0 + ln) * K;
    const bf16* arow1 = A  + (size_t)(m0 + 16 + ln) * K;
    const bf16* brow0 = Wt + (size_t)(n0 + ln) * K;
    const bf16* brow1 = Wt + (size_t)(n0 + 16 + ln) * K;

    v8f acc00 = {}, acc01 = {}, acc10 = {}, acc11 = {};
    for (int kk = 0; kk < K; kk += 32) {
        const v16bf a0 = load_a_frag(arow0, kk, hi);
        const v16bf a1 = load_a_frag(arow1, kk, hi);
        const v16bf b0 = load_b_frag(brow0, kk, hi);
        const v16bf b1 = load_b_frag(brow1, kk, hi);
        acc00 = wmma_bf16(a0, b0, acc00);
        acc01 = wmma_bf16(a0, b1, acc01);
        acc10 = wmma_bf16(a1, b0, acc10);
        acc11 = wmma_bf16(a1, b1, acc11);
    }

    const float bv0 = bias ? bias[n0 + ln] : 0.0f;
    const float bv1 = bias ? bias[n0 + 16 + ln] : 0.0f;

#pragma unroll
    for (int r = 0; r < 8; ++r) {
        const int row0 = m0 + r + 8 * hi;
        const int row1 = m0 + 16 + r + 8 * hi;
        const float v00 = (acc00[r] + bv0) * alpha;
        const float v01 = (acc01[r] + bv1) * alpha;
        const float v10 = (acc10[r] + bv0) * alpha;
        const float v11 = (acc11[r] + bv1) * alpha;
        const size_t i00 = (size_t)row0 * N + n0 + ln;
        const size_t i01 = (size_t)row0 * N + n0 + 16 + ln;
        const size_t i10 = (size_t)row1 * N + n0 + ln;
        const size_t i11 = (size_t)row1 * N + n0 + 16 + ln;
        if (outf) { outf[i00] = v00; outf[i01] = v01; outf[i10] = v10; outf[i11] = v11; }
        if (outb) { outb[i00] = (bf16)v00; outb[i01] = (bf16)v01;
                    outb[i10] = (bf16)v10; outb[i11] = (bf16)v11; }
    }
}

// ---------------------------------------------------------------------------
// Softmax pass A: per-row running max (m) and sum-exp (l) over all S keys.
// One wave handles 16 query rows of one (b,h). Scores via WMMA (DK=64 -> 2 MACs).
// ---------------------------------------------------------------------------
__global__ void softmax_stats(const bf16* __restrict__ Qb, const bf16* __restrict__ Kb,
                              float* __restrict__ Mbuf, float* __restrict__ Lbuf) {
    const int lane = threadIdx.x, ln = lane & 15, hi = lane >> 4;
    const int bh = blockIdx.y, b = bh / H_, h = bh % H_;
    const int q0 = (blockIdx.x * blockDim.y + threadIdx.y) * 16;

    const bf16* qrow = Qb + ((size_t)b * S_ + q0 + ln) * D_ + h * DK_;
    const v16bf aq0 = load_a_frag(qrow, 0, hi);
    const v16bf aq1 = load_a_frag(qrow, 32, hi);

    float m[8], l[8];
#pragma unroll
    for (int r = 0; r < 8; ++r) { m[r] = -1e30f; l[r] = 0.0f; }

    for (int kt = 0; kt < S_ / 16; ++kt) {
        const bf16* krow = Kb + ((size_t)b * S_ + kt * 16 + ln) * D_ + h * DK_;
        v8f acc = {};
        acc = wmma_bf16(aq0, load_b_frag(krow, 0, hi), acc);
        acc = wmma_bf16(aq1, load_b_frag(krow, 32, hi), acc);
#pragma unroll
        for (int r = 0; r < 8; ++r) {
            const float s  = acc[r];
            const float nm = fmaxf(m[r], s);
            l[r] = l[r] * __expf(m[r] - nm) + __expf(s - nm);
            m[r] = nm;
        }
    }

    // Reduce across the 16 lanes holding different key columns (stay in half-wave).
#pragma unroll
    for (int off = 1; off < 16; off <<= 1) {
#pragma unroll
        for (int r = 0; r < 8; ++r) {
            const float mo = __shfl_xor(m[r], off, 32);
            const float lo = __shfl_xor(l[r], off, 32);
            const float nm = fmaxf(m[r], mo);
            l[r] = l[r] * __expf(m[r] - nm) + lo * __expf(mo - nm);
            m[r] = nm;
        }
    }

    if (ln == 0) {
#pragma unroll
        for (int r = 0; r < 8; ++r) {
            const size_t idx = (size_t)bh * S_ + q0 + r + 8 * hi;
            Mbuf[idx] = m[r];
            Lbuf[idx] = l[r];
        }
    }
}

// ---------------------------------------------------------------------------
// Softmax pass B: recompute scores, write normalized P (fp32) to attn_weights,
// and accumulate O = P @ V via WMMA. P tiles staged through LDS to convert the
// C/D accumulator layout into an A-fragment layout (cross-lane transpose).
// ---------------------------------------------------------------------------
__global__ void attn_pv(const bf16* __restrict__ Qb, const bf16* __restrict__ Kb,
                        const bf16* __restrict__ Vt,
                        const float* __restrict__ Mbuf, const float* __restrict__ Lbuf,
                        float* __restrict__ attnw, bf16* __restrict__ attnC) {
    __shared__ __attribute__((aligned(32))) bf16 Pt[4][16 * 32];

    const int lane = threadIdx.x, ln = lane & 15, hi = lane >> 4;
    const int w  = threadIdx.y;
    const int bh = blockIdx.y, b = bh / H_, h = bh % H_;
    const int q0 = (blockIdx.x * blockDim.y + w) * 16;

    const bf16* qrow = Qb + ((size_t)b * S_ + q0 + ln) * D_ + h * DK_;
    const v16bf aq0 = load_a_frag(qrow, 0, hi);
    const v16bf aq1 = load_a_frag(qrow, 32, hi);

    float mrow[8], linv[8];
#pragma unroll
    for (int r = 0; r < 8; ++r) {
        const size_t idx = (size_t)bh * S_ + q0 + r + 8 * hi;
        mrow[r] = Mbuf[idx];
        linv[r] = 1.0f / Lbuf[idx];
    }

    v8f o[4] = {};

    for (int kc = 0; kc < S_ / 32; ++kc) {
#pragma unroll
        for (int sub = 0; sub < 2; ++sub) {
            const int key0 = kc * 32 + sub * 16;
            const bf16* krow = Kb + ((size_t)b * S_ + key0 + ln) * D_ + h * DK_;
            v8f acc = {};
            acc = wmma_bf16(aq0, load_b_frag(krow, 0, hi), acc);
            acc = wmma_bf16(aq1, load_b_frag(krow, 32, hi), acc);
#pragma unroll
            for (int r = 0; r < 8; ++r) {
                const int row = r + 8 * hi;
                const float p = __expf(acc[r] - mrow[r]) * linv[r];
                attnw[((size_t)bh * S_ + q0 + row) * S_ + key0 + ln] = p;
                Pt[w][row * 32 + sub * 16 + ln] = (bf16)p;
            }
        }
        __syncthreads();

        // A-fragment of P (16 q-rows x 32 keys) from LDS
        v16bf ap;
        {
            const bf16* prow = &Pt[w][ln * 32];
            const v8bf lo = *(const v8bf*)(prow + 8 * hi);
            const v8bf hc = *(const v8bf*)(prow + 16 + 8 * hi);
#pragma unroll
            for (int i = 0; i < 8; ++i) { ap[i] = lo[i]; ap[i + 8] = hc[i]; }
        }

#pragma unroll
        for (int t = 0; t < 4; ++t) {
            const bf16* vrow = Vt + ((size_t)bh * DK_ + t * 16 + ln) * S_ + kc * 32;
            o[t] = wmma_bf16(ap, load_b_frag(vrow, 0, hi), o[t]);
        }
        __syncthreads();
    }

    // Write O (already normalized) into concat-head layout [B,S,D] as bf16
#pragma unroll
    for (int t = 0; t < 4; ++t) {
#pragma unroll
        for (int r = 0; r < 8; ++r) {
            const int row = q0 + r + 8 * hi;
            attnC[((size_t)b * S_ + row) * D_ + h * DK_ + t * 16 + ln] = (bf16)o[t][r];
        }
    }
}

// ---------------------------------------------------------------------------
// Host orchestration
// ---------------------------------------------------------------------------
extern "C" void kernel_launch(void* const* d_in, const int* in_sizes, int n_in,
                              void* d_out, int out_size, void* d_ws, size_t ws_size,
                              hipStream_t stream) {
    (void)in_sizes; (void)n_in; (void)out_size; (void)ws_size;

    const float* values  = (const float*)d_in[0];
    const float* keys    = (const float*)d_in[1];
    const float* queries = (const float*)d_in[2];
    const float* Wq = (const float*)d_in[3];  const float* bq = (const float*)d_in[4];
    const float* Wk = (const float*)d_in[5];  const float* bk = (const float*)d_in[6];
    const float* Wv = (const float*)d_in[7];  const float* bv = (const float*)d_in[8];
    const float* Wo = (const float*)d_in[9];  const float* bo = (const float*)d_in[10];

    float* out   = (float*)d_out;
    float* attnw = out + (size_t)B_ * S_ * D_;   // [B,H,S,S] fp32

    const size_t BSD = (size_t)B_ * S_ * D_;
    const size_t DD  = (size_t)D_ * D_;
    const size_t BHS = (size_t)B_ * H_ * S_;

    char* ws = (char*)d_ws;
    size_t off = 0;
    auto alloc = [&](size_t bytes) -> void* {
        void* p = ws + off;
        off = (off + bytes + 255) & ~(size_t)255;
        return p;
    };

    bf16* WqT = (bf16*)alloc(DD * 2);
    bf16* WkT = (bf16*)alloc(DD * 2);
    bf16* WvT = (bf16*)alloc(DD * 2);
    bf16* WoT = (bf16*)alloc(DD * 2);
    bf16* Xq  = (bf16*)alloc(BSD * 2);
    bf16* Xk  = (bf16*)alloc(BSD * 2);
    bf16* Xv  = (bf16*)alloc(BSD * 2);
    bf16* Qb  = (bf16*)alloc(BSD * 2);
    bf16* Kb  = (bf16*)alloc(BSD * 2);
    bf16* Vb  = (bf16*)alloc(BSD * 2);
    bf16* Vt  = (bf16*)alloc(BSD * 2);
    float* Mb = (float*)alloc(BHS * 4);
    float* Lb = (float*)alloc(BHS * 4);
    bf16* attnC = (bf16*)alloc(BSD * 2);
    bf16* attn1 = (bf16*)alloc(BSD * 2);

    // 1) convert inputs + transpose-convert weights to bf16
    cvt_f32_bf16<<<1024, 256, 0, stream>>>(queries, Xq, BSD);
    cvt_f32_bf16<<<1024, 256, 0, stream>>>(keys,    Xk, BSD);
    cvt_f32_bf16<<<1024, 256, 0, stream>>>(values,  Xv, BSD);
    transpose_w_bf16<<<1024, 256, 0, stream>>>(Wq, WqT, D_);
    transpose_w_bf16<<<1024, 256, 0, stream>>>(Wk, WkT, D_);
    transpose_w_bf16<<<1024, 256, 0, stream>>>(Wv, WvT, D_);
    transpose_w_bf16<<<1024, 256, 0, stream>>>(Wo, WoT, D_);

    // 2) Q/K/V projections (1/sqrt(DK)=0.125 folded into Q)
    // 32x32 wave tiles: grid.x over N/32, grid.y over M/(32*4 waves)
    const dim3 gblk(32, 4);
    const dim3 ggrid(D_ / 32, (B_ * S_) / 128);
    gemm_bf16<<<ggrid, gblk, 0, stream>>>(Xq, WqT, bq, 0.125f, Qb, nullptr, B_ * S_, D_, D_);
    gemm_bf16<<<ggrid, gblk, 0, stream>>>(Xk, WkT, bk, 1.0f,  Kb, nullptr, B_ * S_, D_, D_);
    gemm_bf16<<<ggrid, gblk, 0, stream>>>(Xv, WvT, bv, 1.0f,  Vb, nullptr, B_ * S_, D_, D_);

    // 3) transpose V per head: [B,S,D] -> [B,H,DK,S]
    v_transpose<<<1024, 256, 0, stream>>>(Vb, Vt);

    // 4) softmax statistics (pass A), then P + P@V (pass B)
    const dim3 ablk(32, 4);
    const dim3 agrid((S_ / 16) / 4, B_ * H_);
    softmax_stats<<<agrid, ablk, 0, stream>>>(Qb, Kb, Mb, Lb);
    attn_pv<<<agrid, ablk, 0, stream>>>(Qb, Kb, Vt, Mb, Lb, attnw, attnC);

    // 5) output projection applied twice (faithful to reference)
    gemm_bf16<<<ggrid, gblk, 0, stream>>>(attnC, WoT, bo, 1.0f, attn1, nullptr, B_ * S_, D_, D_);
    gemm_bf16<<<ggrid, gblk, 0, stream>>>(attn1, WoT, bo, 1.0f, nullptr, out,  B_ * S_, D_, D_);
}
